// GraphAttentionLayer_83150566851117
// MI455X (gfx1250) — matrix operand order repaired and checked by reading, not compiled, not run on hardware
//
#include <hip/hip_runtime.h>
#include <hip/hip_bf16.h>

typedef __attribute__((ext_vector_type(2)))  float  v2f;
typedef __attribute__((ext_vector_type(8)))  float  v8f;
typedef __attribute__((ext_vector_type(16))) __bf16 v16bf;

#define BS   8
#define NN   2048
#define FIN  128
#define FOUT 64
#define GAT_ALPHA 0.1f

__device__ __forceinline__ float lrelu(float x) { return fmaxf(x, GAT_ALPHA * x); }

// ---------------------------------------------------------------------------
// Kernel 1: Wh = h @ W + b   (fp32 WMMA 16x16x4), also emit WhT in bf16
// (o-major: WhT[b][o][n]) so the attention kernel's B tiles are contiguous.
// One wave per 16x16 output tile: 1024 i-tiles * 4 n-tiles = 4096 waves.
// ---------------------------------------------------------------------------
__global__ __launch_bounds__(256) void k_wh(const float* __restrict__ h,
                                            const float* __restrict__ W,
                                            const float* __restrict__ bias,
                                            float* __restrict__ Wh,
                                            __bf16* __restrict__ WhT) {
  const int lane = threadIdx.x & 31;
  const int wid  = (blockIdx.x * blockDim.x + threadIdx.x) >> 5;
  const int it   = wid >> 2;        // 16-row tile over BS*NN rows
  const int nt   = wid & 3;         // 16-col tile over FOUT
  const int i0   = it * 16;
  const int n0   = nt * 16;
  const int half = lane >> 4;
  const int mrow = lane & 15;
  const int n    = n0 + mrow;

  const float* hrow = h + (size_t)(i0 + mrow) * FIN;

  v8f acc = {};
#pragma unroll 4
  for (int k0 = 0; k0 < FIN; k0 += 4) {
    const int ka = k0 + half * 2;
    v2f a, bm;
    a.x  = hrow[ka];
    a.y  = hrow[ka + 1];
    bm.x = W[(size_t)ka * FOUT + n];
    bm.y = W[(size_t)(ka + 1) * FOUT + n];
    acc = __builtin_amdgcn_wmma_f32_16x16x4_f32(false, a, false, bm,
                                                (short)0, acc, false, false);
  }

  const float bn = bias[n];
  const int   bb = i0 >> 11;                       // batch index
  const int  il0 = (i0 & (NN - 1)) + half * 8;     // row-in-batch base

  union { __bf16 e[8]; uint4 u; } pk;
  float vals[8];
#pragma unroll
  for (int r = 0; r < 8; r++) {
    float v = acc[r] + bn;
    vals[r] = v;
    pk.e[r] = (__bf16)v;
  }
#pragma unroll
  for (int r = 0; r < 8; r++)
    Wh[(size_t)(i0 + half * 8 + r) * FOUT + n] = vals[r];

  // WhT[b][n][row]: 8 contiguous bf16 per lane -> one b128 store
  *(uint4*)(WhT + ((size_t)(bb * FOUT + n)) * NN + il0) = pk.u;
}

// ---------------------------------------------------------------------------
// Kernel 2: e1 = Wh @ a1, e2 = Wh @ a2  (one thread per row)
// ---------------------------------------------------------------------------
__global__ __launch_bounds__(256) void k_e(const float* __restrict__ Wh,
                                           const float* __restrict__ aw,
                                           float* __restrict__ e1,
                                           float* __restrict__ e2) {
  const int gi = blockIdx.x * blockDim.x + threadIdx.x;  // 0..BS*NN-1
  const float* row = Wh + (size_t)gi * FOUT;
  float s1 = 0.f, s2 = 0.f;
#pragma unroll
  for (int o = 0; o < FOUT; o++) {
    float v = row[o];
    s1 += v * aw[o];
    s2 += v * aw[FOUT + o];
  }
  e1[gi] = s1;
  e2[gi] = s2;
}

// ---------------------------------------------------------------------------
// Kernel 3: per-batch e2max, then pack {e2_j, exp(e2_j-mx), exp(0.1(e2_j-mx))}
// per column j. LeakyReLU splits exp(e_ij - m_i) into a separable product on
// each branch, so these 2*N exps replace the N*N dense exps entirely.
// ---------------------------------------------------------------------------
__global__ __launch_bounds__(256) void k_prep(const float* __restrict__ e2,
                                              float* __restrict__ e2max,
                                              float4* __restrict__ pk) {
  __shared__ float red[256];
  const int bb = blockIdx.x, tid = threadIdx.x;
  float mx = -3.4e38f;
  for (int j = tid; j < NN; j += 256) mx = fmaxf(mx, e2[bb * NN + j]);
  red[tid] = mx;
  __syncthreads();
  for (int s = 128; s > 0; s >>= 1) {
    if (tid < s) red[tid] = fmaxf(red[tid], red[tid + s]);
    __syncthreads();
  }
  const float m = red[0];
  if (tid == 0) e2max[bb] = m;
  for (int j = tid; j < NN; j += 256) {
    float v = e2[bb * NN + j];
    float d = v - m;                       // <= 0
    pk[bb * NN + j] = make_float4(v, __expf(d), __expf(0.1f * d), 0.f);
  }
}

// ---------------------------------------------------------------------------
// Kernel 4: fused flash-GAT. One wave owns a 16-row i-tile; streams j in
// chunks of 32. Per element (exact, no exp in the hot loop):
//   z>0 ? A_i*Ep_j : B_i*En_j  ==  exp(lrelu(e1_i+e2_j+ab) - m_i)
// denom += p (dense); A-tile = bf16(p * adj);
// D += A @ WhT-tile via v_wmma_f32_16x16x32_bf16 (4 n-tiles, Fout=64).
// ---------------------------------------------------------------------------
__global__ __launch_bounds__(128) void k_attn(const float* __restrict__ adj,
                                              const __bf16* __restrict__ WhT,
                                              const float* __restrict__ e1g,
                                              const float4* __restrict__ pkg,
                                              const float* __restrict__ e2maxg,
                                              const float* __restrict__ abg,
                                              float* __restrict__ out) {
  __shared__ float4 pks[NN];                // {e2, Ep, En, 0} : 32 KB
  const int bb  = blockIdx.x >> 5;          // 32 blocks per batch
  const int tb  = (blockIdx.x & 31) * 4;    // first i-tile of this block
  const int tid = threadIdx.x;

  for (int q = tid; q < NN; q += 128) pks[q] = pkg[bb * NN + q];
  __syncthreads();

  const int wave = tid >> 5;
  const int lane = tid & 31;
  const int half = lane >> 4;
  const int mrow = lane & 15;
  const int i0   = (tb + wave) * 16;        // row tile within batch
  const int irow = i0 + mrow;               // A-matrix row for this lane

  const float ab    = abg[0];
  const float e1v   = e1g[bb * NN + irow];
  const float zc    = e1v + ab + e2maxg[bb];     // row-max argument
  const float mmax  = lrelu(zc);                 // exact row max of e
  const float Aiv   = __expf(zc - mmax);         // <= 1
  const float Biv   = __expf(0.1f * zc - mmax);  // <= 1
  const float theta = -(e1v + ab);               // branch threshold on e2_j

  const float*  adjrow = adj + ((size_t)(bb * NN + irow)) * NN;
  const __bf16* whtB   = WhT + (size_t)bb * FOUT * NN;

  v8f   acc[4] = {v8f{}, v8f{}, v8f{}, v8f{}};
  float dsum   = 0.f;

  for (int j0 = 0; j0 < NN; j0 += 32) {
    if (j0 + 64 < NN) __builtin_prefetch(adjrow + j0 + 64, 0, 0);

    const int jb0 = j0 + half * 8;
    // adj for this lane's 16 (i,j) entries (A-layout: K = {0..7} U {16..23})
    float4 ad0 = *(const float4*)(adjrow + jb0);
    float4 ad1 = *(const float4*)(adjrow + jb0 + 4);
    float4 ad2 = *(const float4*)(adjrow + jb0 + 16);
    float4 ad3 = *(const float4*)(adjrow + jb0 + 20);
    float adv[16] = {ad0.x, ad0.y, ad0.z, ad0.w, ad1.x, ad1.y, ad1.z, ad1.w,
                     ad2.x, ad2.y, ad2.z, ad2.w, ad3.x, ad3.y, ad3.z, ad3.w};

    v16bf amat;
#pragma unroll
    for (int e = 0; e < 16; e++) {
      const int j = jb0 + ((e < 8) ? e : (e + 8));
      float4 q   = pks[j];                  // broadcast b128 LDS read
      bool   pos = (q.x > theta);
      float  p   = (pos ? q.y : q.z) * (pos ? Aiv : Biv);
      dsum      += p;                       // dense softmax denominator
      amat[e]    = (__bf16)(p * adv[e]);    // masked numerator (bf16 A)
    }

    // B tiles: WhT rows are contiguous in K; lane-half h takes K = {8h..8h+7} U {16+8h..23+8h}
    const int kb = j0 + half * 8;
#pragma unroll
    for (int t = 0; t < 4; t++) {
      union { uint4 u[2]; v16bf v; } bu;
      const __bf16* bp = whtB + (size_t)(t * 16 + mrow) * NN + kb;
      bu.u[0] = *(const uint4*)(bp);
      bu.u[1] = *(const uint4*)(bp + 16);
      acc[t] = __builtin_amdgcn_wmma_f32_16x16x32_bf16(false, amat, false, bu.v,
                                                       (short)0, acc[t],
                                                       false, false);
    }
  }

  // denom per row: row q lives in lanes q and q+16
  float tot = dsum + __shfl_xor(dsum, 16, 32);   // lane L holds denom[L&15]

#pragma unroll
  for (int r = 0; r < 8; r++) {
    const int m = half * 8 + r;                  // C-layout row
    float rinv  = 1.0f / __shfl(tot, m, 32);
    float* op = out + ((size_t)(bb * NN + i0 + m)) * FOUT + mrow;
    op[0]  = acc[0][r] * rinv;
    op[16] = acc[1][r] * rinv;
    op[32] = acc[2][r] * rinv;
    op[48] = acc[3][r] * rinv;
  }
}

// ---------------------------------------------------------------------------
extern "C" void kernel_launch(void* const* d_in, const int* in_sizes, int n_in,
                              void* d_out, int out_size, void* d_ws, size_t ws_size,
                              hipStream_t stream) {
  const float* h   = (const float*)d_in[0];
  const float* adj = (const float*)d_in[1];
  const float* W   = (const float*)d_in[2];
  const float* bv  = (const float*)d_in[3];
  const float* aw  = (const float*)d_in[4];
  const float* ab  = (const float*)d_in[5];
  float* out = (float*)d_out;

  char* ws = (char*)d_ws;
  float*  Wh    = (float*)ws;                                   // 4 MB
  __bf16* WhT   = (__bf16*)(ws + 4u * 1024 * 1024);             // 2 MB
  float*  e1    = (float*)(ws + 6u * 1024 * 1024);              // 64 KB
  float*  e2    = (float*)(ws + 6u * 1024 * 1024 + 65536);      // 64 KB
  float*  e2max = (float*)(ws + 6u * 1024 * 1024 + 2 * 65536);  // 32 B
  float4* pk    = (float4*)(ws + 6u * 1024 * 1024 + 3 * 65536); // 256 KB

  k_wh<<<dim3(512), dim3(256), 0, stream>>>(h, W, bv, Wh, WhT);
  k_e<<<dim3((BS * NN) / 256), dim3(256), 0, stream>>>(Wh, aw, e1, e2);
  k_prep<<<dim3(BS), dim3(256), 0, stream>>>(e2, e2max, pk);
  k_attn<<<dim3(256), dim3(128), 0, stream>>>(adj, WhT, e1, pk, e2max, ab, out);
}
